// CompILE_20933670601161
// MI455X (gfx1250) — compile-verified, matrix operand order checked by reading, not run on hardware
//
#include <hip/hip_runtime.h>
#include <hip/hip_bf16.h>

// ---------------------------------------------------------------------------
// Problem constants (from the reference)
// ---------------------------------------------------------------------------
static constexpr int B = 256, T = 128, V = 512, H = 256, L = 64, S = 4;
static constexpr float EPSC = 1e-17f;
static constexpr float NEG_INF = -1e30f;

typedef __bf16 bf16_t;
typedef __attribute__((ext_vector_type(16))) __bf16 v16bf;
typedef __attribute__((ext_vector_type(8)))  __bf16 v8bf;
typedef __attribute__((ext_vector_type(8)))  float  v8f;

// ---------------------------------------------------------------------------
// WMMA helpers (CDNA5 wave32, V_WMMA_F32_16X16X32_BF16)
// A fragment (16x32, MxK): lane m = lane&15, half hf = lane>>4.
//   VGPR v holds K pair k = kbase + hf*8 + ((v&4)<<2) + ((v&3)<<1)  -> two
//   contiguous 8-element runs at kbase+hf*8 and kbase+16+hf*8.
// B fragment (32x16, KxN): lane n = lane&15, same K striping (k-major from
//   row-major weight W[N][K], i.e. column n of W^T).
// ---------------------------------------------------------------------------
__device__ __forceinline__ v16bf load_frag(const float* __restrict__ row, int kbase) {
  int hf = (threadIdx.x >> 4) & 1;
  const float* r = row + kbase + hf * 8;
  v16bf out;
#pragma unroll
  for (int v = 0; v < 8; ++v) {
    int k = ((v & 4) << 2) + ((v & 3) << 1);  // 0,2,4,6,16,18,20,22
    out[2 * v]     = (bf16_t)r[k];
    out[2 * v + 1] = (bf16_t)r[k + 1];
  }
  return out;
}

// bf16 source (global or LDS): two contiguous 16B chunks per fragment.
__device__ __forceinline__ v16bf load_frag_bf(const bf16_t* __restrict__ row, int kbase) {
  int hf = (threadIdx.x >> 4) & 1;
  union { v8bf h[2]; v16bf w; } u;
  u.h[0] = *(const v8bf*)(row + kbase + hf * 8);
  u.h[1] = *(const v8bf*)(row + kbase + 16 + hf * 8);
  return u.w;
}

__device__ __forceinline__ v8f wmma_bf16(v16bf a, v16bf b, v8f c) {
  return __builtin_amdgcn_wmma_f32_16x16x32_bf16(
      false, a, false, b, (short)0, c, false, false);
}

__device__ __forceinline__ float sigf(float x) { return 1.0f / (1.0f + __expf(-x)); }

// splitmix64 -> uniform in (0,1]
__device__ __forceinline__ float uhash(unsigned a, unsigned b, unsigned c) {
  unsigned long long x = ((unsigned long long)a << 42) ^ ((unsigned long long)b << 21)
                       ^ (unsigned long long)c;
  x += 0x9E3779B97F4A7C15ull;
  x = (x ^ (x >> 30)) * 0xBF58476D1CE4E5B9ull;
  x = (x ^ (x >> 27)) * 0x94D049BB133111EBull;
  x ^= x >> 31;
  return (float)((double)((x >> 11) + 1) * 1.1102230246251565e-16); // (0,1]
}

// ---------------------------------------------------------------------------
// f32 -> bf16 conversion (weight pre-pass)
// ---------------------------------------------------------------------------
__global__ __launch_bounds__(256) void k_cvt_bf16(
    const float* __restrict__ src, bf16_t* __restrict__ dst, int n) {
  int i = blockIdx.x * 256 + threadIdx.x;
  if (i < n) dst[i] = (bf16_t)src[i];
}

// ---------------------------------------------------------------------------
// xgates[t][b][4H] = embed_w[inputs[b*T+t]] @ w_ih^T + (b_ih + b_hh)
// One wave per 16x16 tile of [B*T, 4H]; output re-laid out time-major so the
// recurrent kernel streams one contiguous slab per step.
// ---------------------------------------------------------------------------
__global__ __launch_bounds__(128) void k_xgates(
    const float* __restrict__ embed_w, const int* __restrict__ inputs,
    const float* __restrict__ w_ih, const float* __restrict__ b_ih,
    const float* __restrict__ b_hh, float* __restrict__ xg) {
  int wave = threadIdx.x >> 5;
  int tile = blockIdx.x * 4 + wave;              // 2048 * 64 tiles
  int mt = tile >> 6, nt = tile & 63;
  int lane = threadIdx.x & 31;
  int l15 = lane & 15, hf = lane >> 4;

  int m = mt * 16 + l15;                         // row in [0, B*T)
  int n = nt * 16 + l15;                         // col in [0, 4H)
  const float* arow = embed_w + (size_t)inputs[m] * H;
  const float* brow = w_ih + (size_t)n * H;

  v8f acc = {};
#pragma unroll
  for (int kb = 0; kb < H; kb += 32)
    acc = wmma_bf16(load_frag(arow, kb), load_frag(brow, kb), acc);

  float bias = b_ih[n] + b_hh[n];
#pragma unroll
  for (int r = 0; r < 8; ++r) {
    int row = mt * 16 + r + 8 * hf;              // = b*T + t
    int b = row >> 7, t = row & (T - 1);
    xg[((size_t)t * B + b) * (4 * H) + n] = acc[r] + bias;
  }
}

// ---------------------------------------------------------------------------
// Persistent masked-LSTM encode for one segment.
// Grid: 16 blocks (one 16-row batch strip each), 512 threads = 16 waves.
// Wave w owns H-columns [16w,16w+16): computes the 4 gate tiles via WMMA,
// keeps c in registers, h as bf16 in LDS, and -- since w_hh is
// time-invariant -- its weight fragments (32 x v16bf = 256 VGPRs) in
// registers across the entire time loop.
// ---------------------------------------------------------------------------
__global__ __launch_bounds__(512) void k_lstm_seg(
    const float* __restrict__ xg, const bf16_t* __restrict__ whh_bf,
    const float* __restrict__ mask, float* __restrict__ enc) {
  constexpr int HP = H + 8;                      // padded bf16 row stride (16B pad)
  __shared__ bf16_t hsh[16 * HP];

  int wave = threadIdx.x >> 5;                   // n-tile this wave owns
  int lane = threadIdx.x & 31;
  int l15 = lane & 15, hf = lane >> 4;
  int b0 = blockIdx.x * 16;
  int col = wave * 16 + l15;

  for (int i = threadIdx.x; i < 16 * HP; i += 512) hsh[i] = (bf16_t)0.f;

  // hoist the time-invariant recurrent weights into registers
  v16bf wfr[4][8];
#pragma unroll
  for (int g = 0; g < 4; ++g) {
    const bf16_t* brow = whh_bf + (size_t)(g * H + col) * H;
#pragma unroll
    for (int kk = 0; kk < 8; ++kk) wfr[g][kk] = load_frag_bf(brow, kk * 32);
  }

  float creg[8];
#pragma unroll
  for (int r = 0; r < 8; ++r) creg[r] = 0.f;

  const bf16_t* arow = hsh + l15 * HP;
  __syncthreads();

#pragma unroll 1
  for (int t = 0; t < T; ++t) {
    // prefetch next step's x-gate slab while the WMMAs run
    if (t + 1 < T)
      __builtin_prefetch(xg + ((size_t)(t + 1) * B + b0 + 8 * hf) * (4 * H) + col, 0, 1);

    v8f acc[4];
#pragma unroll
    for (int g = 0; g < 4; ++g) acc[g] = (v8f){};

#pragma unroll
    for (int kk = 0; kk < 8; ++kk) {
      v16bf a = load_frag_bf(arow, kk * 32);
#pragma unroll
      for (int g = 0; g < 4; ++g)
        acc[g] = wmma_bf16(a, wfr[g][kk], acc[g]);
    }
    __syncthreads();                             // all reads of h done

#pragma unroll
    for (int r = 0; r < 8; ++r) {
      int b = b0 + r + 8 * hf;
      const float* xrow = xg + ((size_t)t * B + b) * (4 * H);
      float iv = acc[0][r] + xrow[col];
      float fv = acc[1][r] + xrow[H + col];
      float gv = acc[2][r] + xrow[2 * H + col];
      float ov = acc[3][r] + xrow[3 * H + col];
      float c2 = sigf(fv) * creg[r] + sigf(iv) * tanhf(gv);
      float h2 = sigf(ov) * tanhf(c2);
      float mm = mask[(size_t)b * T + t];
      h2 *= mm; c2 *= mm;
      creg[r] = c2;
      enc[((size_t)b * T + t) * H + col] = h2;
      hsh[(r + 8 * hf) * HP + col] = (bf16_t)h2;
    }
    __syncthreads();                             // h visible for next step
  }
}

// ---------------------------------------------------------------------------
// Boundary head, fused:  lb[bt] = relu(enc[bt]@wb1^T + bb1) . wb2 + bb2
// ---------------------------------------------------------------------------
__global__ __launch_bounds__(128) void k_boundary(
    const float* __restrict__ enc, const float* __restrict__ wb1,
    const float* __restrict__ bb1, const float* __restrict__ wb2,
    const float* __restrict__ bb2, float* __restrict__ lb) {
  int wave = threadIdx.x >> 5;
  int mt = blockIdx.x * 4 + wave;                // 2048 row-strips
  int lane = threadIdx.x & 31;
  int l15 = lane & 15, hf = lane >> 4;

  const float* arow = enc + (size_t)(mt * 16 + l15) * H;
  v16bf afr[8];
#pragma unroll
  for (int kk = 0; kk < 8; ++kk) afr[kk] = load_frag(arow, kk * 32);

  float part[8];
#pragma unroll
  for (int r = 0; r < 8; ++r) part[r] = 0.f;

  for (int nt = 0; nt < 16; ++nt) {
    int n = nt * 16 + l15;
    const float* brow = wb1 + (size_t)n * H;
    v8f acc = {};
#pragma unroll
    for (int kk = 0; kk < 8; ++kk)
      acc = wmma_bf16(afr[kk], load_frag(brow, kk * 32), acc);
    float w2 = wb2[n], b1 = bb1[n];
#pragma unroll
    for (int r = 0; r < 8; ++r) {
      float hid = fmaxf(acc[r] + b1, 0.f);
      part[r] += hid * w2;
    }
  }
#pragma unroll
  for (int r = 0; r < 8; ++r) {
    float v = part[r];
#pragma unroll
    for (int off = 1; off < 16; off <<= 1) v += __shfl_xor(v, off);
    part[r] = v;
  }
  if (l15 == 0) {
    float b2 = bb2[0];
#pragma unroll
    for (int r = 0; r < 8; ++r)
      lb[mt * 16 + r + 8 * hf] = part[r] + b2;
  }
}

// ---------------------------------------------------------------------------
// Gumbel-softmax over T per row (block = one row, 128 threads)
// ---------------------------------------------------------------------------
__global__ __launch_bounds__(128) void k_gumbel_softmax(
    const float* __restrict__ lb_in, float* __restrict__ b_logits_out,
    float* __restrict__ b_samples_out, float* __restrict__ sb_ws, int seg) {
  __shared__ float red[128];
  int b = blockIdx.x, t = threadIdx.x;
  float lb = (t == 0) ? NEG_INF : lb_in[b * T + t];
  b_logits_out[b * T + t] = lb;
  float u = uhash(2u * seg, (unsigned)b, (unsigned)t);
  float gum = -__logf(EPSC - __logf(u + EPSC));
  float x = lb + gum;                            // TEMP_B == 1
  red[t] = x; __syncthreads();
  for (int s = 64; s > 0; s >>= 1) { if (t < s) red[t] = fmaxf(red[t], red[t + s]); __syncthreads(); }
  float mx = red[0]; __syncthreads();
  float e = __expf(x - mx);
  red[t] = e; __syncthreads();
  for (int s = 64; s > 0; s >>= 1) { if (t < s) red[t] += red[t + s]; __syncthreads(); }
  float sm = e / red[0];
  b_samples_out[b * T + t] = sm;
  sb_ws[b * T + t] = sm;
}

__global__ __launch_bounds__(128) void k_onehot(
    const int* __restrict__ lengths, float* __restrict__ b_samples_out,
    float* __restrict__ sb_ws) {
  int b = blockIdx.x, t = threadIdx.x;
  float v = (t == lengths[b] - 1) ? 1.f : 0.f;
  b_samples_out[b * T + t] = v;
  sb_ws[b * T + t] = v;
}

// readout[b,h] = sum_{t<T-1} enc[b,t,h] * sb[b,t+1]
__global__ __launch_bounds__(256) void k_readout(
    const float* __restrict__ enc, const float* __restrict__ sb,
    float* __restrict__ ro) {
  int idx = blockIdx.x * 256 + threadIdx.x;      // B*H
  int b = idx >> 8, h = idx & 255;
  float acc = 0.f;
  for (int t = 0; t < T - 1; ++t)
    acc += enc[((size_t)b * T + t) * H + h] * sb[b * T + t + 1];
  ro[idx] = acc;
}

// generic small dense layer: out[m,n] = act(bias[n] + in[m,:K].W[n,:K])
__global__ __launch_bounds__(256) void k_linear(
    const float* __restrict__ in, const float* __restrict__ W,
    const float* __restrict__ bias, float* __restrict__ out,
    int M, int N, int K, int relu) {
  int idx = blockIdx.x * 256 + threadIdx.x;
  if (idx >= M * N) return;
  int m = idx / N, n = idx - m * N;
  const float* a = in + (size_t)m * K;
  const float* w = W + (size_t)n * K;
  float acc = bias[n];
  for (int k = 0; k < K; ++k) acc = fmaf(a[k], w[k], acc);
  if (relu) acc = fmaxf(acc, 0.f);
  out[idx] = acc;
}

// z = mu + exp(0.5*logvar) * N(0,1)   (Box-Muller from hash)
__global__ __launch_bounds__(256) void k_zsample(
    const float* __restrict__ lz, float* __restrict__ z_out,
    float* __restrict__ z_ws, int seg) {
  int idx = blockIdx.x * 256 + threadIdx.x;      // B*L
  int b = idx / L, l = idx - b * L;
  float mu = lz[b * 2 * L + l];
  float lv = lz[b * 2 * L + L + l];
  float u1 = uhash(2u * seg + 1u, (unsigned)b, (unsigned)l);
  float u2 = uhash(2u * seg + 1u, (unsigned)b + 4096u, (unsigned)l);
  float eps = __fsqrt_rn(-2.f * __logf(u1)) * __cosf(6.2831853f * u2);
  float z = mu + __expf(0.5f * lv) * eps;
  z_out[idx] = z;
  z_ws[idx] = z;
}

// per-row cumsum of sample_b, accumulate log, emit mask
__global__ __launch_bounds__(128) void k_cummask(
    const float* __restrict__ sb, float* __restrict__ log_acc,
    float* __restrict__ mask_ws, float* __restrict__ masks_out) {
  __shared__ float s[128];
  int b = blockIdx.x, t = threadIdx.x;
  s[t] = sb[b * T + t]; __syncthreads();
  for (int off = 1; off < 128; off <<= 1) {
    float v = (t >= off) ? s[t - off] : 0.f;
    __syncthreads();
    s[t] += v;
    __syncthreads();
  }
  float acc = log_acc[b * T + t] + __logf(s[t] + EPSC);
  log_acc[b * T + t] = acc;
  float m = __expf(acc);
  mask_ws[b * T + t] = m;
  masks_out[b * T + t] = m;
}

// recs[seg][b][t][v] = pred[b][v]   (pure bandwidth, float4)
__global__ __launch_bounds__(256) void k_broadcast_rec(
    const float* __restrict__ pred, float* __restrict__ rec) {
  size_t idx = ((size_t)blockIdx.x * 256 + threadIdx.x) * 4;   // over B*T*V
  size_t v = idx & (V - 1);
  size_t b = idx >> 16;                                        // / (T*V)
  float4 p = *(const float4*)(pred + b * V + v);
  *(float4*)(rec + idx) = p;
}

__global__ __launch_bounds__(256) void k_fill(float* __restrict__ p, float v, int n) {
  int i = blockIdx.x * 256 + threadIdx.x;
  if (i < n) p[i] = v;
}

// ---------------------------------------------------------------------------
// Host orchestration
// ---------------------------------------------------------------------------
extern "C" void kernel_launch(void* const* d_in, const int* in_sizes, int n_in,
                              void* d_out, int out_size, void* d_ws, size_t ws_size,
                              hipStream_t stream) {
  const float* embed_w = (const float*)d_in[0];
  const float* w_ih    = (const float*)d_in[1];
  const float* w_hh    = (const float*)d_in[2];
  const float* b_ih    = (const float*)d_in[3];
  const float* b_hh    = (const float*)d_in[4];
  const float* wz1     = (const float*)d_in[5];
  const float* bz1     = (const float*)d_in[6];
  const float* wz2     = (const float*)d_in[7];
  const float* bz2     = (const float*)d_in[8];
  const float* wb1     = (const float*)d_in[9];
  const float* bb1     = (const float*)d_in[10];
  const float* wb2     = (const float*)d_in[11];
  const float* bb2     = (const float*)d_in[12];
  const float* wd1     = (const float*)d_in[13];
  const float* bd1     = (const float*)d_in[14];
  const float* wd2     = (const float*)d_in[15];
  const float* bd2     = (const float*)d_in[16];
  const int*   inputs  = (const int*)d_in[17];
  const int*   lengths = (const int*)d_in[18];

  float* out = (float*)d_out;
  const size_t off_encs     = 0;
  const size_t off_recs     = off_encs     + (size_t)S * B * T * H;
  const size_t off_masks    = off_recs     + (size_t)S * B * T * V;
  const size_t off_blogits  = off_masks    + (size_t)(S - 1) * B * T;
  const size_t off_bsamples = off_blogits  + (size_t)(S - 1) * B * T;
  const size_t off_zlogits  = off_bsamples + (size_t)S * B * T;
  const size_t off_zsamples = off_zlogits  + (size_t)S * B * 2 * L;

  float* ws = (float*)d_ws;
  float* xg      = ws;                          // [T][B][4H] f32
  bf16_t* whh_bf = (bf16_t*)(xg + (size_t)B * T * 4 * H);  // 4H*H bf16
  float* mask_ws = (float*)(whh_bf + (size_t)4 * H * H);   // B*T
  float* log_acc = mask_ws + B * T;
  float* lb_ws   = log_acc + B * T;
  float* sb_ws   = lb_ws + B * T;
  float* ro_ws   = sb_ws + B * T;               // B*H
  float* hz_ws   = ro_ws + B * H;               // B*H
  float* z_ws    = hz_ws + B * H;               // B*L
  float* hd_ws   = z_ws + B * L;                // B*H
  float* pred_ws = hd_ws + B * H;               // B*V

  // init mask = 1, log_cum_acc = 0; pre-convert w_hh to bf16
  k_fill<<<(B * T + 255) / 256, 256, 0, stream>>>(mask_ws, 1.0f, B * T);
  k_fill<<<(B * T + 255) / 256, 256, 0, stream>>>(log_acc, 0.0f, B * T);
  k_cvt_bf16<<<(4 * H * H + 255) / 256, 256, 0, stream>>>(w_hh, whh_bf, 4 * H * H);

  // precompute x-projection of all LSTM gates once (segment-invariant)
  k_xgates<<<(B * T / 16) * (4 * H / 16) / 4, 128, 0, stream>>>(
      embed_w, inputs, w_ih, b_ih, b_hh, xg);

  for (int seg = 0; seg < S; ++seg) {
    float* enc_seg = out + off_encs + (size_t)seg * B * T * H;

    // masked LSTM encode: one persistent kernel, whole time loop inside
    k_lstm_seg<<<B / 16, 512, 0, stream>>>(xg, whh_bf, mask_ws, enc_seg);

    // boundary sample
    if (seg < S - 1) {
      k_boundary<<<(B * T / 16) / 4, 128, 0, stream>>>(
          enc_seg, wb1, bb1, wb2, bb2, lb_ws);
      k_gumbel_softmax<<<B, T, 0, stream>>>(
          lb_ws, out + off_blogits + (size_t)seg * B * T,
          out + off_bsamples + (size_t)seg * B * T, sb_ws, seg);
    } else {
      k_onehot<<<B, T, 0, stream>>>(
          lengths, out + off_bsamples + (size_t)seg * B * T, sb_ws);
    }

    // latent z
    k_readout<<<B * H / 256, 256, 0, stream>>>(enc_seg, sb_ws, ro_ws);
    k_linear<<<(B * H + 255) / 256, 256, 0, stream>>>(
        ro_ws, wz1, bz1, hz_ws, B, H, H, 1);
    float* zlog = out + off_zlogits + (size_t)seg * B * 2 * L;
    k_linear<<<(B * 2 * L + 255) / 256, 256, 0, stream>>>(
        hz_ws, wz2, bz2, zlog, B, 2 * L, H, 0);
    k_zsample<<<(B * L + 255) / 256, 256, 0, stream>>>(
        zlog, out + off_zsamples + (size_t)seg * B * L, z_ws, seg);

    // next-segment mask
    if (seg < S - 1) {
      k_cummask<<<B, T, 0, stream>>>(
          sb_ws, log_acc, mask_ws, out + off_masks + (size_t)seg * B * T);
    }

    // decode + broadcast
    k_linear<<<(B * H + 255) / 256, 256, 0, stream>>>(
        z_ws, wd1, bd1, hd_ws, B, H, L, 1);
    k_linear<<<(B * V + 255) / 256, 256, 0, stream>>>(
        hd_ws, wd2, bd2, pred_ws, B, V, H, 0);
    k_broadcast_rec<<<(B * T * V / 4) / 256, 256, 0, stream>>>(
        pred_ws, out + off_recs + (size_t)seg * B * T * V);
  }
  (void)in_sizes; (void)n_in; (void)out_size; (void)ws_size;
}
